// RowAttentionBlock_3539053052297
// MI455X (gfx1250) — matrix operand order repaired and checked by reading, not compiled, not run on hardware
//
#include <hip/hip_runtime.h>

typedef __attribute__((ext_vector_type(16))) __bf16 v16bf;
typedef __attribute__((ext_vector_type(8)))  __bf16 v8bf;
typedef __attribute__((ext_vector_type(4)))  __bf16 v4bf;
typedef __attribute__((ext_vector_type(8)))  float  v8f;
typedef __attribute__((ext_vector_type(4)))  float  v4f;
typedef __attribute__((ext_vector_type(4)))  unsigned int v4u;
typedef __attribute__((ext_vector_type(8)))  int    v8i;
typedef __attribute__((ext_vector_type(4)))  int    v4i;
typedef unsigned int  u32;
typedef unsigned short u16;

#define FDIM   128
#define PITCH  136   // LDS row pitch in bf16 elements (= 256B row + 16B pad)
#define ROWS   262144  // 512*512 total (p,q) positions

#if defined(__gfx1250__) && __has_builtin(__builtin_amdgcn_tensor_load_to_lds) && \
    __has_builtin(__builtin_amdgcn_s_wait_tensorcnt)
#define USE_TDM 1
#else
#define USE_TDM 0
#endif

union BFrag { uint4 q[2]; v16bf v; };
union Cvt8  { v8bf b; uint4 u; };
union Cvt4  { v4bf b; uint2 u; };

// Packed f32 -> bf16 via hardware v_cvt_pk_bf16_f32, RNE.
__device__ __forceinline__ uint4 pack8v(v8f v) {
    Cvt8 c; c.b = __builtin_convertvector(v, v8bf);
    return c.u;
}
__device__ __forceinline__ uint2 pack4v(v4f v) {
    Cvt4 c; c.b = __builtin_convertvector(v, v4bf);
    return c.u;
}

// A-fragment (16x32 bf16): lane row = lm; 16 contiguous K at f0+16*hi
__device__ __forceinline__ BFrag load_afrag(const u16* base, int row, int pitch, int f0, int hi) {
    BFrag fr;
    const u16* p0 = base + row * pitch + f0 + hi * 16;
    fr.q[0] = *(const uint4*)p0;
    fr.q[1] = *(const uint4*)(p0 + 8);
    return fr;
}
// B-fragment (32x16 bf16): lane col = lm; 16 contiguous K at f0+16*hi
__device__ __forceinline__ BFrag load_bfrag(const u16* base, int row, int pitch, int f0, int hi) {
    BFrag fr;
    const u16* p0 = base + row * pitch + f0 + hi * 16;
    fr.q[0] = *(const uint4*)p0;
    fr.q[1] = *(const uint4*)(p0 + 8);
    return fr;
}

#define WMMA_BF16(A, B, C) \
    __builtin_amdgcn_wmma_f32_16x16x32_bf16(false, (A), false, (B), (short)0, (C), false, false)

#if USE_TDM
// Tensor Data Mover: DMA a 128x128 bf16 tile (row stride = row_stride elems) from
// global into LDS at lds_byte_addr, inserting 16B padding every 256B row so the
// LDS layout lands at PITCH=136 elements per row. D# per CDNA5 ISA ch.8.
// This toolchain exposes the 6-arg builtin: (v4u g0, v8i g1, v4i g2, v4i g3, v8i, i32 cpol).
__device__ __forceinline__ void tdm_load_tile(u32 lds_byte_addr, const void* gptr,
                                              u32 row_stride) {
    unsigned long long ga = (unsigned long long)gptr;
    v4u g0;
    g0[0] = 1u;                                        // count=1 valid descriptor
    g0[1] = lds_byte_addr;                             // lds_addr (bytes)
    g0[2] = (u32)ga;                                   // global_addr[31:0]
    g0[3] = (u32)((ga >> 32) & 0x1FFFFFFu) | (2u << 30); // addr[56:32] | type=2
    v8i g1;
    g1[0] = (1 << 16)        // data_size = 1 -> 2 bytes
          | (1 << 20)        // pad_enable
          | (5 << 22)        // pad_interval: 2^(5+1)=64 DWORDs = 256B
          | (3 << 25);       // pad_amount: 3+1 = 4 DWORDs = 16B
    g1[1] = (128 << 16);     // atomic_barrier_addr=0 | tensor_dim0 lo16 = 128
    g1[2] = (128 << 16);     // tensor_dim0 hi16 = 0 | tensor_dim1 lo16 = 128
    g1[3] = (128 << 16);     // tensor_dim1 hi16 = 0 | tile_dim0 = 128
    g1[4] = 128;             // tile_dim1 = 128 | tile_dim2 = 0
    g1[5] = (int)row_stride; // tensor_dim0_stride lo32 (data_size units)
    g1[6] = 0;               // stride hi16 | tensor_dim1_stride lo16
    g1[7] = 0;               // tensor_dim1_stride hi32
    v4i z4 = { 0, 0, 0, 0 };
    v8i z8 = { 0, 0, 0, 0, 0, 0, 0, 0 };
    __builtin_amdgcn_tensor_load_to_lds(g0, g1, z4, z4, z8, 0);
}
#endif

// ---------------------------------------------------------------------------
// Kernel 1: fused RMSNorm + Q/K/V projections.
// Grid: 2048 WGs x 256 threads; each WG handles 128 (p,q) rows.
// Q, K stored row-major bf16 [row][g]; V stored transposed bf16 Vt[p][g][n].
// ---------------------------------------------------------------------------
__global__ __launch_bounds__(256) void qkv_kernel(
    const float* __restrict__ x, const float* __restrict__ nscale,
    const float* __restrict__ pwq, const float* __restrict__ pwk,
    const float* __restrict__ pwv, const float* __restrict__ pbv,
    u16* __restrict__ Q, u16* __restrict__ K, u16* __restrict__ Vt)
{
    __shared__ alignas(16) u16 h[128 * PITCH];        // normalized activations, bf16
    __shared__ alignas(16) u16 w[3][128 * PITCH];     // wq, wk, wv in bf16

    const int tid  = threadIdx.x;
    const int lane = tid & 31;
    const int wid  = tid >> 5;
    const int hi   = lane >> 4;
    const int lm   = lane & 15;
    const long rowbase = (long)blockIdx.x * 128;

    // ---- Phase A: RMSNorm -> h (bf16, LDS). Wave handles 16 rows, fully coalesced.
    float4 nsv = ((const float4*)nscale)[lane];
    for (int rr = 0; rr < 16; ++rr) {
        int row = wid * 16 + rr;
        float4 xv = ((const float4*)(x + (rowbase + row) * FDIM))[lane];
        float ss = xv.x * xv.x + xv.y * xv.y + xv.z * xv.z + xv.w * xv.w;
        ss += __shfl_xor(ss, 1, 32);  ss += __shfl_xor(ss, 2, 32);
        ss += __shfl_xor(ss, 4, 32);  ss += __shfl_xor(ss, 8, 32);
        ss += __shfl_xor(ss, 16, 32);
        float rs = rsqrtf(ss * (1.0f / 128.0f) + 1e-6f);
        v4f hv = { xv.x * rs * nsv.x, xv.y * rs * nsv.y,
                   xv.z * rs * nsv.z, xv.w * rs * nsv.w };
        *(uint2*)&h[row * PITCH + lane * 4] = pack4v(hv);
    }

    // ---- Phase B: weights fp32 -> bf16 LDS (L2-resident, coalesced).
    const float* wm[3] = { pwq, pwk, pwv };
    for (int m = 0; m < 3; ++m) {
        for (int i = tid; i < 128 * 128 / 4; i += 256) {
            float4 v = ((const float4*)wm[m])[i];
            int row = (i * 4) >> 7, col = (i * 4) & 127;
            v4f wv4 = { v.x, v.y, v.z, v.w };
            *(uint2*)&w[m][row * PITCH + col] = pack4v(wv4);
        }
    }
    __syncthreads();

    // ---- Phase C1: Q, K computed transposed: out^T = W x h^T.
    // A = weight rows (wave owns g-strip [16*wid,16*wid+16)), B = h columns (queries).
    for (int mat = 0; mat < 2; ++mat) {
        u16* dst = (mat == 0) ? Q : K;
        const u16* W = w[mat];
        BFrag afr[4];
        #pragma unroll
        for (int s = 0; s < 4; ++s) afr[s] = load_afrag(W, wid * 16 + lm, PITCH, s * 32, hi);
        #pragma unroll
        for (int t = 0; t < 8; ++t) {
            int qr = t * 16 + lm;          // B column = query row in h
            v8f acc = { 0, 0, 0, 0, 0, 0, 0, 0 };
            #pragma unroll
            for (int s = 0; s < 4; ++s) {
                BFrag bf = load_bfrag(h, qr, PITCH, s * 32, hi);
                acc = WMMA_BF16(afr[s].v, bf.v, acc);
            }
            // element: row q = rowbase+t*16+lm, cols g = wid*16 + hi*8 + r (contiguous)
            *(uint4*)&dst[(rowbase + t * 16 + lm) * FDIM + wid * 16 + hi * 8] = pack8v(acc);
        }
    }

    // ---- Phase C2: V computed normally (A = h rows), stored transposed Vt[p][g][n].
    {
        BFrag afr[4];
        #pragma unroll
        for (int s = 0; s < 4; ++s) afr[s] = load_afrag(h, wid * 16 + lm, PITCH, s * 32, hi);
        const long p  = blockIdx.x >> 2;
        const int  n0 = (blockIdx.x & 3) * 128;
        #pragma unroll
        for (int t = 0; t < 8; ++t) {
            int g = t * 16 + lm;           // B column = output feature
            float bias = pbv[g];
            v8f acc = { 0, 0, 0, 0, 0, 0, 0, 0 };
            #pragma unroll
            for (int s = 0; s < 4; ++s) {
                BFrag bf = load_bfrag(w[2], g, PITCH, s * 32, hi);
                acc = WMMA_BF16(afr[s].v, bf.v, acc);
            }
            #pragma unroll
            for (int r = 0; r < 8; ++r) acc[r] += bias;
            // element: key n = n0 + wid*16 + hi*8 + r (contiguous), feature g
            *(uint4*)&Vt[(p * 128 + g) * 512 + n0 + wid * 16 + hi * 8] = pack8v(acc);
        }
    }
}

// ---------------------------------------------------------------------------
// Kernel 2: per-row flash attention. Grid: 2048 WGs (512 rows x 4 query tiles).
// Wave owns 16 queries; computes S^T = K_chunk x Q^T so each lane owns one
// query column -> softmax reduction is one shfl_xor(16). O^T = V^T x P^T.
// K/V chunks staged into LDS by the Tensor Data Mover (with HW row padding).
// ---------------------------------------------------------------------------
__global__ __launch_bounds__(256) void attn_kernel(
    const u16* __restrict__ Q, const u16* __restrict__ K,
    const u16* __restrict__ Vt, float* __restrict__ out)
{
    __shared__ alignas(16) u16 Ks[128 * PITCH];      // key chunk, row-major [n][f]
    __shared__ alignas(16) u16 Vs[128 * PITCH];      // V^T chunk, [g][n]
    __shared__ alignas(16) u16 Pb[8][16 * PITCH];    // per-wave P, row-major [m][n]

    const int tid  = threadIdx.x;
    const int lane = tid & 31;
    const int wid  = tid >> 5;
    const int hi   = lane >> 4;
    const int lm   = lane & 15;
    const long p   = blockIdx.x >> 2;
    const int  q0  = (blockIdx.x & 3) * 128;
    const long qrow = p * 512 + q0 + wid * 16 + lm;   // this lane's query
    const float scale = 0.08838834764831843f;          // 1/sqrt(128)

    // Q fragments (B operand of S^T), live in registers for the whole kernel.
    BFrag qf[4];
    #pragma unroll
    for (int s = 0; s < 4; ++s) {
        const u16* p0 = Q + qrow * FDIM + s * 32 + hi * 16;
        qf[s].q[0] = *(const uint4*)p0;
        qf[s].q[1] = *(const uint4*)(p0 + 8);
    }

    v8f oacc[8];
    #pragma unroll
    for (int t = 0; t < 8; ++t) oacc[t] = (v8f){ 0, 0, 0, 0, 0, 0, 0, 0 };
    float mrun = -1e30f, lrun = 0.0f;

    for (int c = 0; c < 4; ++c) {                      // 4 key chunks of 128
        __syncthreads();                               // prior chunk fully consumed
        const u16* Kg = K + (p * 512 + c * 128) * FDIM;
        const u16* Vg = Vt + p * 128 * 512 + c * 128;
#if USE_TDM
        if (wid == 0) {
            tdm_load_tile((u32)(unsigned long long)(uintptr_t)&Ks[0], Kg, FDIM);
            tdm_load_tile((u32)(unsigned long long)(uintptr_t)&Vs[0], Vg, 512);
            __builtin_amdgcn_s_wait_tensorcnt(0);
        }
#else
        for (int i = tid; i < 2048; i += 256) {        // 128*128 bf16 / 8-per-thread
            int r = i >> 4, c8 = (i & 15) * 8;
            *(uint4*)&Ks[r * PITCH + c8] = *(const uint4*)&Kg[r * FDIM + c8];
        }
        for (int i = tid; i < 2048; i += 256) {
            int g = i >> 4, c8 = (i & 15) * 8;
            *(uint4*)&Vs[g * PITCH + c8] = *(const uint4*)&Vg[g * 512 + c8];
        }
#endif
        __syncthreads();

        // S^T tiles: rows = keys (8 tiles of 16), cols = wave's 16 queries.
        v8f sacc[8];
        #pragma unroll
        for (int t = 0; t < 8; ++t) {
            v8f a = { 0, 0, 0, 0, 0, 0, 0, 0 };
            #pragma unroll
            for (int s = 0; s < 4; ++s) {
                BFrag kf = load_afrag(Ks, t * 16 + lm, PITCH, s * 32, hi);
                a = WMMA_BF16(kf.v, qf[s].v, a);
            }
            sacc[t] = a;
        }

        // Online softmax (per query = per lane column).
        float mx = -1e30f;
        #pragma unroll
        for (int t = 0; t < 8; ++t)
            #pragma unroll
            for (int r = 0; r < 8; ++r) mx = fmaxf(mx, sacc[t][r]);
        mx = fmaxf(mx, __shfl_xor(mx, 16, 32));
        float mnew = fmaxf(mrun, mx * scale);
        float corr = __expf(mrun - mnew);
        float psum = 0.0f;
        #pragma unroll
        for (int t = 0; t < 8; ++t) {
            v8f pv;
            #pragma unroll
            for (int r = 0; r < 8; ++r) {
                pv[r] = __expf(sacc[t][r] * scale - mnew);
                psum += pv[r];
            }
            // lane owns (query lm, keys t*16+hi*8 .. +8) -> row-major P, b128 store
            *(uint4*)&Pb[wid][lm * PITCH + t * 16 + hi * 8] = pack8v(pv);
        }
        psum += __shfl_xor(psum, 16, 32);
        lrun = lrun * corr + psum;
        mrun = mnew;
        #pragma unroll
        for (int t = 0; t < 8; ++t)
            #pragma unroll
            for (int r = 0; r < 8; ++r) oacc[t][r] *= corr;

        // P fragments (B operand of O^T): wave-private, dscnt-ordered.
        BFrag pf[4];
        #pragma unroll
        for (int s = 0; s < 4; ++s) {
            const u16* p0 = &Pb[wid][lm * PITCH + s * 32 + hi * 16];
            pf[s].q[0] = *(const uint4*)p0;
            pf[s].q[1] = *(const uint4*)(p0 + 8);
        }
        // O^T += V^T x P^T : rows = features g (8 tiles), cols = queries.
        #pragma unroll
        for (int t = 0; t < 8; ++t) {
            v8f a = oacc[t];
            #pragma unroll
            for (int s = 0; s < 4; ++s) {
                BFrag vf = load_afrag(Vs, t * 16 + lm, PITCH, s * 32, hi);
                a = WMMA_BF16(vf.v, pf[s].v, a);
            }
            oacc[t] = a;
        }
    }

    // Normalize and store: lane holds (query qrow, features t*16+hi*8 .. +8) fp32.
    float inv = 1.0f / lrun;
    #pragma unroll
    for (int t = 0; t < 8; ++t) {
        float4 o0 = { oacc[t][0] * inv, oacc[t][1] * inv, oacc[t][2] * inv, oacc[t][3] * inv };
        float4 o1 = { oacc[t][4] * inv, oacc[t][5] * inv, oacc[t][6] * inv, oacc[t][7] * inv };
        float* dst = out + qrow * FDIM + t * 16 + hi * 8;
        ((float4*)dst)[0] = o0;
        ((float4*)dst)[1] = o1;
    }
}

// ---------------------------------------------------------------------------
extern "C" void kernel_launch(void* const* d_in, const int* in_sizes, int n_in,
                              void* d_out, int out_size, void* d_ws, size_t ws_size,
                              hipStream_t stream) {
    const float* x   = (const float*)d_in[0];
    const float* ns  = (const float*)d_in[1];
    const float* wq  = (const float*)d_in[2];
    const float* wk  = (const float*)d_in[3];
    const float* wv  = (const float*)d_in[4];
    const float* bv  = (const float*)d_in[5];
    float* out = (float*)d_out;

    // Workspace: Q, K (row-major) + Vt (transposed), bf16: 3 * 64 MB = 192 MB.
    u16* Q  = (u16*)d_ws;
    u16* K  = Q + (size_t)ROWS * FDIM;
    u16* Vt = K + (size_t)ROWS * FDIM;

    qkv_kernel<<<2048, 256, 0, stream>>>(x, ns, wq, wk, wv, bv, Q, K, Vt);
    attn_kernel<<<2048, 256, 0, stream>>>(Q, K, Vt, out);
}
